// OutputConvNew_88880053223996
// MI455X (gfx1250) — matrix operand order repaired and checked by reading, not compiled, not run on hardware
//
#include <hip/hip_runtime.h>

typedef __attribute__((ext_vector_type(16))) _Float16 v16h;
typedef __attribute__((ext_vector_type(8)))  _Float16 v8h;
typedef __attribute__((ext_vector_type(8)))  float    v8f;

// Problem constants (from reference)
#define NB   48
#define CIN  64
#define HH   128
#define WW   128
#define C1   128
#define C2   32

// ---------------------------------------------------------------------------
// pack_x: NCHW f32 -> NHWC f16
// ---------------------------------------------------------------------------
__global__ void pack_x_kernel(const float* __restrict__ src, _Float16* __restrict__ dst) {
    size_t idx = (size_t)blockIdx.x * blockDim.x + threadIdx.x;
    int x  = (int)(idx % WW);
    int y  = (int)((idx / WW) % HH);
    int ch = (int)((idx / ((size_t)WW * HH)) % CIN);
    int n  = (int)(idx / ((size_t)WW * HH * CIN));
    dst[(((size_t)n * HH + y) * WW + x) * CIN + ch] = (_Float16)src[idx];
}

// ---------------------------------------------------------------------------
// pack_w: OIHW f32 -> per-fragment swizzled f16 B blocks.
// Fragment (r,s,c,nt) is 512 halves; lane l slot j holds
//   B[k = c*32 + 16*(l>=16) + j][n = nt*16 + (l&15)]  (32x16 K-major B layout)
// ---------------------------------------------------------------------------
__global__ void pack_w_kernel(const float* __restrict__ Wsrc, _Float16* __restrict__ dst,
                              int Cin, int Cout) {
    int idx = blockIdx.x * blockDim.x + threadIdx.x;
    int KC = Cin / 32, NT = Cout / 16;
    int total = 9 * KC * NT * 512;
    if (idx >= total) return;
    int j    = idx & 15;
    int lane = (idx >> 4) & 31;
    int frag = idx >> 9;
    int nt = frag % NT; frag /= NT;
    int c  = frag % KC; frag /= KC;
    int r  = frag / 3, s = frag % 3;
    int h  = lane >> 4, nn = lane & 15;
    int k  = c * 32 + 16 * h + j;
    int o  = nt * 16 + nn;
    dst[idx] = (_Float16)Wsrc[((o * Cin + k) * 3 + r) * 3 + s];
}

// ---------------------------------------------------------------------------
// Implicit-GEMM 3x3 SAME conv, WMMA f16 -> f32 accum, LDS halo staging.
//
// Block strip: 16*MTB consecutive pixels of one image row; all Cout channels.
// Waves/block = (Cout/16) * (MTB/MTW); each wave owns one 16-out-channel tile
// and MTW pixel sub-tiles.
//   - LDS halo is zero-padded at staging time -> unconditional ds_load_b128.
//   - Per-pixel LDS stride padded to Cin+8 halves: lane-m bank start is
//     (36*m) mod 64, 16 distinct 4-bank spans tiling all 64 banks -> no
//     bank conflicts on A-fragment reads.
//   - All MTW A-fragments are loaded before the MTW WMMAs: one ds clause,
//     one s_wait_dscnt, then back-to-back v_wmma with distinct D regs.
// ---------------------------------------------------------------------------
template <int Cin, int Cout, int MTB, int MTW, bool RELU>
__global__ void conv3x3_wmma_kernel(const _Float16* __restrict__ in,
                                    const _Float16* __restrict__ wp,
                                    const float* __restrict__ bias,
                                    _Float16* __restrict__ out) {
    constexpr int KC     = Cin / 32;
    constexpr int NT     = Cout / 16;
    constexpr int STRIP  = 16 * MTB;        // pixels per block
    constexpr int STRIPP = STRIP + 2;       // with 1-pixel halo each side
    constexpr int XT     = WW / STRIP;      // strips per row
    constexpr int ROWQ   = Cin + 8;         // padded per-pixel LDS stride (halves)

    __shared__ _Float16 halo[3 * STRIPP * ROWQ];

    // Block -> (n, y, x0)
    int bt = blockIdx.x;
    int xt = bt % XT;
    int y  = (bt / XT) % HH;
    int n  = bt / (XT * HH);
    int x0 = xt * STRIP;

    int tid  = threadIdx.x;
    int lane = tid & 31;
    int wid  = tid >> 5;
    int nt   = wid % NT;                    // out-channel tile
    int mg   = wid / NT;                    // pixel-group within strip
    int m    = lane & 15;
    int h    = lane >> 4;

    // ---- Stage zero-padded halo into LDS (8 halves per thread-chunk) ----
    constexpr int CV = Cin / 8;
    for (int i = tid; i < 3 * STRIPP * CV; i += (int)blockDim.x) {
        int cv = i % CV;
        int pc = (i / CV) % STRIPP;
        int rr = i / (CV * STRIPP);
        int yy = y + rr - 1;
        int xx = x0 + pc - 1;
        v8h v = {};
        if (yy >= 0 && yy < HH && xx >= 0 && xx < WW)
            v = *(const v8h*)(in + (((size_t)n * HH + yy) * WW + xx) * Cin + cv * 8);
        *(v8h*)(&halo[((size_t)rr * STRIPP + pc) * ROWQ + cv * 8]) = v;
    }
    __syncthreads();

    // ---- WMMA main loop ----
    v8f acc[MTW];
#pragma unroll
    for (int t = 0; t < MTW; ++t) acc[t] = (v8f){};

#pragma unroll
    for (int r = 0; r < 3; ++r) {
#pragma unroll
        for (int s = 0; s < 3; ++s) {
#pragma unroll
            for (int c = 0; c < KC; ++c) {
                // B fragment: one contiguous 32B/lane block, reused MTW times
                const _Float16* wf =
                    wp + ((size_t)(((r * 3 + s) * KC + c) * NT + nt)) * 512 + lane * 16;
                v16h b = *(const v16h*)wf;

                // Load all MTW A-fragments first (single ds clause + wait)
                v16h afrag[MTW];
#pragma unroll
                for (int t = 0; t < MTW; ++t) {
                    int px = (mg * MTW + t) * 16 + m + s;   // halo col (x0-1 origin)
                    const _Float16* ap = &halo[((size_t)r * STRIPP + px) * ROWQ + c * 32];
                    union { v16h v; v8h p[2]; } u;
                    u.p[0] = *(const v8h*)(ap + 16 * h);       // K = 16h .. 16h+7
                    u.p[1] = *(const v8h*)(ap + 16 * h + 8);   // K = 16h+8 .. 16h+15
                    afrag[t] = u.v;
                }
                // Back-to-back WMMAs, distinct accumulators (no D->A/B RAW)
#pragma unroll
                for (int t = 0; t < MTW; ++t) {
                    acc[t] = __builtin_amdgcn_wmma_f32_16x16x32_f16(
                        false, afrag[t], false, b, (short)0, acc[t], false, false);
                }
            }
        }
    }

    // ---- Bias + activation + store (C/D layout: N = lane%16, M = i + 8h) ----
    int ch = nt * 16 + m;
    float bv = bias[ch];
#pragma unroll
    for (int t = 0; t < MTW; ++t) {
        size_t obase = ((size_t)n * HH + y) * WW + x0 + (mg * MTW + t) * 16;
#pragma unroll
        for (int i = 0; i < 8; ++i) {
            int mm = i + 8 * h;
            float v = acc[t][i] + bv;
            if (RELU) v = v > 0.f ? v : 0.f;
            out[(obase + mm) * Cout + ch] = (_Float16)v;
        }
    }
}

// ---------------------------------------------------------------------------
// Fused half-pixel bilinear x2 upsample + 3x3 conv (32 -> 1), zero pad.
// ---------------------------------------------------------------------------
__global__ void conv3_up_kernel(const _Float16* __restrict__ y2,
                                const float* __restrict__ W3,
                                const float* __restrict__ b3,
                                float* __restrict__ out) {
    __shared__ float w3s[C2 * 9];
    for (int i = threadIdx.x; i < C2 * 9; i += blockDim.x) w3s[i] = W3[i];
    __syncthreads();

    const int HO = 2 * HH, WO = 2 * WW;
    size_t gp = (size_t)blockIdx.x * blockDim.x + threadIdx.x;
    int ox = (int)(gp % WO);
    int oy = (int)((gp / WO) % HO);
    int n  = (int)(gp / ((size_t)WO * HO));

    float acc = b3[0];
#pragma unroll
    for (int r = 0; r < 3; ++r) {
        int uy = oy - 1 + r;
        if (uy < 0 || uy >= HO) continue;       // conv zero padding
        int my = uy >> 1, oddy = uy & 1;
        int ya = oddy ? my : my - 1;
        int yb = oddy ? my + 1 : my;
        float wya = oddy ? 0.75f : 0.25f;
        float wyb = 1.0f - wya;
        ya = ya < 0 ? 0 : (ya > HH - 1 ? HH - 1 : ya);  // resize edge clamp
        yb = yb < 0 ? 0 : (yb > HH - 1 ? HH - 1 : yb);
#pragma unroll
        for (int s = 0; s < 3; ++s) {
            int ux = ox - 1 + s;
            if (ux < 0 || ux >= WO) continue;
            int mx = ux >> 1, oddx = ux & 1;
            int xa = oddx ? mx : mx - 1;
            int xb = oddx ? mx + 1 : mx;
            float wxa = oddx ? 0.75f : 0.25f;
            float wxb = 1.0f - wxa;
            xa = xa < 0 ? 0 : (xa > WW - 1 ? WW - 1 : xa);
            xb = xb < 0 ? 0 : (xb > WW - 1 ? WW - 1 : xb);
            const _Float16* p00 = y2 + (((size_t)n * HH + ya) * WW + xa) * C2;
            const _Float16* p01 = y2 + (((size_t)n * HH + ya) * WW + xb) * C2;
            const _Float16* p10 = y2 + (((size_t)n * HH + yb) * WW + xa) * C2;
            const _Float16* p11 = y2 + (((size_t)n * HH + yb) * WW + xb) * C2;
#pragma unroll
            for (int cc = 0; cc < C2 / 8; ++cc) {
                v8h a00 = *(const v8h*)(p00 + cc * 8);
                v8h a01 = *(const v8h*)(p01 + cc * 8);
                v8h a10 = *(const v8h*)(p10 + cc * 8);
                v8h a11 = *(const v8h*)(p11 + cc * 8);
#pragma unroll
                for (int e = 0; e < 8; ++e) {
                    float v = wya * (wxa * (float)a00[e] + wxb * (float)a01[e]) +
                              wyb * (wxa * (float)a10[e] + wxb * (float)a11[e]);
                    acc += v * w3s[(cc * 8 + e) * 9 + r * 3 + s];
                }
            }
        }
    }
    out[gp] = acc;
}

// ---------------------------------------------------------------------------
// Host-side launch
// ---------------------------------------------------------------------------
extern "C" void kernel_launch(void* const* d_in, const int* in_sizes, int n_in,
                              void* d_out, int out_size, void* d_ws, size_t ws_size,
                              hipStream_t stream) {
    (void)in_sizes; (void)n_in; (void)out_size; (void)ws_size;
    const float* x  = (const float*)d_in[0];
    // d_in[1] = to_process (arange, full batch), d_in[2] = batch_size: unused
    const float* W1 = (const float*)d_in[3];
    const float* b1 = (const float*)d_in[4];
    const float* W2 = (const float*)d_in[5];
    const float* b2 = (const float*)d_in[6];
    const float* W3 = (const float*)d_in[7];
    const float* b3 = (const float*)d_in[8];
    float* out = (float*)d_out;

    // Workspace layout (all offsets 256B-aligned)
    const size_t XH_BYTES = (size_t)NB * HH * WW * CIN * 2;  // 100.7 MB
    const size_t Y1_BYTES = (size_t)NB * HH * WW * C1 * 2;   // 201.3 MB
    const size_t W1P_N = 9 * (CIN / 32) * (C1 / 16) * 512;   // 73728 halves
    const size_t W2P_N = 9 * (C1 / 32) * (C2 / 16) * 512;    // 36864 halves

    char* ws = (char*)d_ws;
    _Float16* xh  = (_Float16*)(ws);
    _Float16* y1  = (_Float16*)(ws + XH_BYTES);
    _Float16* y2  = (_Float16*)(ws);  // alias xh: x no longer needed after conv1
    _Float16* w1p = (_Float16*)(ws + XH_BYTES + Y1_BYTES);
    _Float16* w2p = (_Float16*)(ws + XH_BYTES + Y1_BYTES + W1P_N * 2);

    // 1) stage activations + weights in f16
    {
        size_t total = (size_t)NB * CIN * HH * WW;           // 50.3M, /256 exact
        pack_x_kernel<<<(unsigned)(total / 256), 256, 0, stream>>>(x, xh);
    }
    pack_w_kernel<<<(unsigned)((W1P_N + 255) / 256), 256, 0, stream>>>(W1, w1p, CIN, C1);
    pack_w_kernel<<<(unsigned)((W2P_N + 255) / 256), 256, 0, stream>>>(W2, w2p, C1, C2);

    // Strip of 64 pixels per block; blocks = 48*128*(128/64) = 12288
    const unsigned CONV_BLOCKS = NB * HH * (WW / 64);

    // 2) conv1: 64 -> 128, ReLU. waves = 8 (NT=8, MTW=4 -> 1 m-group), 256 thr
    conv3x3_wmma_kernel<CIN, C1, /*MTB=*/4, /*MTW=*/4, true>
        <<<CONV_BLOCKS, (C1 / 16) * (4 / 4) * 32, 0, stream>>>(xh, w1p, b1, y1);

    // 3) conv2: 128 -> 32, ReLU. waves = 4 (NT=2, MTW=2 -> 2 m-groups), 128 thr
    conv3x3_wmma_kernel<C1, C2, /*MTB=*/4, /*MTW=*/2, true>
        <<<CONV_BLOCKS, (C2 / 16) * (4 / 2) * 32, 0, stream>>>(y1, w2p, b2, y2);

    // 4) fused bilinear x2 + conv3 (32 -> 1). 48*256*256 = 3145728 pixels
    conv3_up_kernel<<<3145728 / 256, 256, 0, stream>>>(y2, W3, b3, out);
}
